// SemanticAlignmentModule_43404939494211
// MI455X (gfx1250) — matrix operand order repaired and checked by reading, not compiled, not run on hardware
//
#include <hip/hip_runtime.h>
#include <hip/hip_bf16.h>
#include <math.h>
#include <stdint.h>

// ---------------------------------------------------------------- types
typedef __attribute__((ext_vector_type(16))) _Float16 v16h;
typedef __attribute__((ext_vector_type(8)))  _Float16 v8h;
typedef __attribute__((ext_vector_type(8)))  float    v8f;

#define BB 8
#define NN 16384          // H*W
#define CV 256
#define CT 512
#define TT 77
#define TP1 80            // padded T for sim (5 x 16)
#define TP2 96            // padded T for attn@v K dim (3 x 32)
#define NHEAD 8
#define HD 32

__device__ inline v8f zero8() {
  v8f z = {0.f,0.f,0.f,0.f,0.f,0.f,0.f,0.f};
  return z;
}

__device__ inline v8f wmma16(v16h a, v16h b, v8f c) {
  // D = A(16x32 f16) * B(32x16 f16) + C(16x16 f32)
  return __builtin_amdgcn_wmma_f32_16x16x32_f16(false, a, false, b, (short)0, c, false, false);
}

// A-fragment (16x32, row-major source with leading dim ld halves, K offset k0)
// lane l<16: row M=l, halves {K=k0..k0+7, K=k0+16..k0+23}; lane l+16: +8 shift
__device__ inline v16h lda(const _Float16* base, int ld, int k0) {
  const int l = threadIdx.x & 31;
  const _Float16* p = base + (l & 15) * ld + k0 + (l >> 4) * 8;
  union { v16h v; v8h h[2]; } u;
  u.h[0] = *(const v8h*)(p);
  u.h[1] = *(const v8h*)(p + 16);
  return u.v;
}

// B-fragment (32x16) from row-major [K][N] (ld halves): lane l = column N=n0+l%16,
// halves {K=k0..k0+7, K=k0+16..k0+23}; lane l+16: +8 shift (matches lda).
__device__ inline v16h ldb(const _Float16* mat, int ld, int k0, int n0) {
  const int l = threadIdx.x & 31;
  const _Float16* p = mat + (size_t)(k0 + (l >> 4) * 8) * ld + n0 + (l & 15);
  v16h v;
#pragma unroll
  for (int s = 0; s < 8; ++s) {
    v[s] = p[(size_t)s * ld];
    v[s + 8] = p[(size_t)(s + 16) * ld];
  }
  return v;
}

// Async global->LDS copy of 16 bytes (per lane), tracked with ASYNCcnt.
__device__ inline void async_g2l_b128(unsigned lds_byte_off, const void* gaddr) {
  asm volatile("global_load_async_to_lds_b128 %0, %1, off"
               :: "v"(lds_byte_off), "v"(gaddr) : "memory");
}
__device__ inline void wait_async0() {
  asm volatile("s_wait_asynccnt 0" ::: "memory");
}

// ---------------------------------------------------------------- kernels
__global__ void cvt_f16(const float* __restrict__ s, _Float16* __restrict__ d, int n) {
  int i = blockIdx.x * 256 + threadIdx.x;
  if (i < n) d[i] = (_Float16)s[i];
}

// One block per (b, t) incl. pad tokens. Computes k = l2norm(text@k_w + k_b)
// stored transposed [b][h][d][TP1] (WMMA-B ready), v = text@v_w+v_b stored
// [b][h][TP2][d], and pad bias (-1e30 for zero tokens / pad slots).
__global__ __launch_bounds__(256) void text_prep(
    const float* __restrict__ text,
    const float* __restrict__ kw, const float* __restrict__ kb,
    const float* __restrict__ vw, const float* __restrict__ vb,
    _Float16* __restrict__ knt, _Float16* __restrict__ vvv, float* __restrict__ bias)
{
  const int b = blockIdx.x / TP2, t = blockIdx.x % TP2;
  const int tid = threadIdx.x;
  const int h = tid >> 5, d = tid & 31;
  if (t >= TT) {                       // pad slots: zero / -inf every call
    vvv[((size_t)(b * NHEAD + h) * TP2 + t) * HD + d] = (_Float16)0.f;
    if (t < TP1) {
      knt[((size_t)(b * NHEAD + h) * HD + d) * TP1 + t] = (_Float16)0.f;
      if (tid == 0) bias[b * TP1 + t] = -1e30f;
    }
    return;
  }
  __shared__ float stx[CT];
  __shared__ float s_as, s_n;
  if (tid == 0) { s_as = 0.f; s_n = 0.f; }
  __syncthreads();
  float as = 0.f;
  for (int i = tid; i < CT; i += 256) {
    float v = text[((size_t)b * TT + t) * CT + i];
    stx[i] = v; as += fabsf(v);
  }
  atomicAdd(&s_as, as);
  __syncthreads();
  const int c = tid;
  float kf = kb[c], vf = vb[c];
  for (int i = 0; i < CT; ++i) {
    float tx = stx[i];
    kf = fmaf(tx, kw[i * CV + c], kf);
    vf = fmaf(tx, vw[i * CV + c], vf);
  }
  atomicAdd(&s_n, kf * kf);
  __syncthreads();
  float inv = 1.f / fmaxf(sqrtf(s_n), 1e-6f);
  knt[((size_t)(b * NHEAD + h) * HD + d) * TP1 + t] = (_Float16)(kf * inv);
  vvv[((size_t)(b * NHEAD + h) * TP2 + t) * HD + d] = (_Float16)vf;
  if (tid == 0) bias[b * TP1 + t] = (s_as <= 1e-6f) ? -1e30f : 0.f;
}

// Fused: LN1 -> q proj (WMMA) -> l2norm -> sim (WMMA) -> top5+softmax ->
// attn@v (WMMA) -> out proj (WMMA) -> gate/residual -> LN2 -> y (d_out).
// Block = 16 rows, 256 threads (8 waves, head h = wave id for attention).
__global__ __launch_bounds__(256) void attn_kernel(
    const float* __restrict__ visual,
    const float* __restrict__ n1w, const float* __restrict__ n1b,
    const _Float16* __restrict__ qwh, const float* __restrict__ qb,
    const _Float16* __restrict__ owh, const float* __restrict__ ob,
    const float* __restrict__ gw, const float* __restrict__ gb,
    const float* __restrict__ lscale, const float* __restrict__ palpha,
    const float* __restrict__ n2w, const float* __restrict__ n2b,
    const _Float16* __restrict__ knt, const _Float16* __restrict__ vvv,
    const float* __restrict__ bias, float* __restrict__ yout)
{
  __shared__ alignas(32) float    sx[16 * CV];        // LN1(x)
  __shared__ alignas(32) _Float16 sxh[16 * CV];       // f16 staging (x, qn, aligned)
  __shared__ alignas(32) float    sq[16 * CV];        // q_full / aligned
  __shared__ alignas(32) float    ssim[16 * NHEAD * TP1]; // sim / out-proj reuse
  __shared__ alignas(32) _Float16 sattn[16 * NHEAD * TP2];
  __shared__ float rs[16][16], rq[16][16];
  __shared__ float smu[16], siv[16], sgate[16];

  const int tid = threadIdx.x, lane = tid & 31, wv = tid >> 5;
  const int sub = tid & 15, row = tid >> 4;
  const int col = lane & 15, hi = lane >> 4;
  const int b  = blockIdx.x >> 10;              // 1024 tiles / batch
  const int m0 = (blockIdx.x & 1023) << 4;
  const size_t gbase = ((size_t)b * NN + m0) * CV;

  // ---- 1. load + LayerNorm1 -------------------------------------------
  float vbuf[16];
  {
    const float* src = visual + gbase + (size_t)row * CV + sub * 16;
    float s1 = 0.f, s2 = 0.f;
#pragma unroll
    for (int j = 0; j < 16; ++j) { float v = src[j]; vbuf[j] = v; s1 += v; s2 += v * v; }
    rs[row][sub] = s1; rq[row][sub] = s2;
  }
  __syncthreads();
  if (tid < 16) {
    float a = 0.f, c2 = 0.f;
#pragma unroll
    for (int j = 0; j < 16; ++j) { a += rs[tid][j]; c2 += rq[tid][j]; }
    float mu = a * (1.f / 256.f);
    smu[tid] = mu; siv[tid] = rsqrtf(c2 * (1.f / 256.f) - mu * mu + 1e-5f);
  }
  __syncthreads();
  {
    float mu = smu[row], iv = siv[row];
#pragma unroll
    for (int j = 0; j < 16; ++j) {
      int c = sub * 16 + j;
      float xn = (vbuf[j] - mu) * iv * n1w[c] + n1b[c];
      sx[row * CV + c] = xn; sxh[row * CV + c] = (_Float16)xn;
    }
  }
  __syncthreads();

  // ---- 2. q = x @ q_w + q_b (WMMA, K=256) ------------------------------
  {
    v16h a[8];
#pragma unroll
    for (int k = 0; k < 8; ++k) a[k] = lda(sxh, CV, k * 32);
#pragma unroll
    for (int it = 0; it < 2; ++it) {
      int nt = wv * 2 + it;
      v8f acc = zero8();
#pragma unroll
      for (int k = 0; k < 8; ++k) acc = wmma16(a[k], ldb(qwh, CV, k * 32, nt * 16), acc);
      float bc = qb[nt * 16 + col];
#pragma unroll
      for (int r = 0; r < 8; ++r) sq[(r + 8 * hi) * CV + nt * 16 + col] = acc[r] + bc;
    }
  }
  __syncthreads();

  // ---- 3. l2-normalize q rows (over full CV), q16 into sxh -------------
  {
    float s = 0.f;
#pragma unroll
    for (int j = 0; j < 16; ++j) { float v = sq[row * CV + sub * 16 + j]; s += v * v; }
    rs[row][sub] = s;
  }
  __syncthreads();
  if (tid < 16) {
    float s = 0.f;
#pragma unroll
    for (int j = 0; j < 16; ++j) s += rs[tid][j];
    siv[tid] = 1.f / fmaxf(sqrtf(s), 1e-6f);
  }
  __syncthreads();
  {
    float iv = siv[row];
#pragma unroll
    for (int j = 0; j < 16; ++j) {
      int c = sub * 16 + j;
      sxh[row * CV + c] = (_Float16)(sq[row * CV + c] * iv);
    }
  }
  __syncthreads();

  // ---- 4. sim = scale * q.k^T + mask (WMMA, one per head x t-tile) -----
  {
    float ls = fminf(fmaxf(lscale[0], -2.f), 2.f);
    float scl = expf(ls) * 0.17677669529663687f;   // / sqrt(32)
    const int h = wv;
    v16h aq = lda(sxh, CV, h * HD);
    const _Float16* kbp = knt + (size_t)(b * NHEAD + h) * HD * TP1;
#pragma unroll
    for (int tt = 0; tt < 5; ++tt) {
      v8f acc = zero8();
      acc = wmma16(aq, ldb(kbp, TP1, 0, tt * 16), acc);
      int t = tt * 16 + col;
      float bv = bias[b * TP1 + t];
#pragma unroll
      for (int r = 0; r < 8; ++r)
        ssim[(r + 8 * hi) * (NHEAD * TP1) + h * TP1 + t] = acc[r] * scl + bv;
    }
  }
  __syncthreads();

  // ---- 5. top-5 threshold + softmax over T (one thread per row,head) ---
  if (tid < 128) {
    const int rr = tid >> 3, h = tid & 7;
    const float* p = &ssim[rr * (NHEAD * TP1) + h * TP1];
    float t0 = -3e38f, t1 = -3e38f, t2 = -3e38f, t3 = -3e38f, t4 = -3e38f;
    for (int t = 0; t < TP1; ++t) {
      float v = p[t];
      if (v > t0)      { t4 = t3; t3 = t2; t2 = t1; t1 = t0; t0 = v; }
      else if (v > t1) { t4 = t3; t3 = t2; t2 = t1; t1 = v; }
      else if (v > t2) { t4 = t3; t3 = t2; t2 = v; }
      else if (v > t3) { t4 = t3; t3 = v; }
      else if (v > t4) { t4 = v; }
    }
    float thr = t4, mx = t0, ssum = 0.f;
    for (int t = 0; t < TP1; ++t) { float v = p[t]; ssum += (v >= thr) ? expf(v - mx) : 0.f; }
    float isum = 1.f / ssum;
    _Float16* ap = &sattn[rr * (NHEAD * TP2) + h * TP2];
    for (int t = 0; t < TP1; ++t) {
      float v = p[t];
      ap[t] = (_Float16)((v >= thr) ? expf(v - mx) * isum : 0.f);
    }
    for (int t = TP1; t < TP2; ++t) ap[t] = (_Float16)0.f;
  }
  __syncthreads();

  // ---- 6. aligned = attn @ v (WMMA, K=96) into sq ----------------------
  {
    const int h = wv;
    const _Float16* ap = &sattn[h * TP2];                       // ld = NHEAD*TP2
    const _Float16* vp = vvv + (size_t)(b * NHEAD + h) * TP2 * HD;
#pragma unroll
    for (int dt = 0; dt < 2; ++dt) {
      v8f acc = zero8();
#pragma unroll
      for (int kk = 0; kk < 3; ++kk) {
        v16h af = lda(ap, NHEAD * TP2, kk * 32);
        acc = wmma16(af, ldb(vp, HD, kk * 32, dt * 16), acc);
      }
#pragma unroll
      for (int r = 0; r < 8; ++r)
        sq[(r + 8 * hi) * CV + h * HD + dt * 16 + col] = acc[r];
    }
  }
  __syncthreads();

  // ---- 7. aligned -> f16 (sxh) ----------------------------------------
#pragma unroll
  for (int j = 0; j < 16; ++j) {
    int c = sub * 16 + j;
    sxh[row * CV + c] = (_Float16)sq[row * CV + c];
  }
  __syncthreads();

  // ---- 8. out proj (WMMA, K=256) into sout (reuse ssim) ----------------
  float* sout = ssim;
  {
    v16h a[8];
#pragma unroll
    for (int k = 0; k < 8; ++k) a[k] = lda(sxh, CV, k * 32);
#pragma unroll
    for (int it = 0; it < 2; ++it) {
      int nt = wv * 2 + it;
      v8f acc = zero8();
#pragma unroll
      for (int k = 0; k < 8; ++k) acc = wmma16(a[k], ldb(owh, CV, k * 32, nt * 16), acc);
      float bc = ob[nt * 16 + col];
#pragma unroll
      for (int r = 0; r < 8; ++r) sout[(r + 8 * hi) * CV + nt * 16 + col] = acc[r] + bc;
    }
  }
  __syncthreads();

  // ---- 9. gate = alpha * sigmoid(x @ gate_w + gate_b) ------------------
  {
    float s = 0.f;
#pragma unroll
    for (int j = 0; j < 16; ++j) { int c = sub * 16 + j; s += sx[row * CV + c] * gw[c]; }
    rs[row][sub] = s;
  }
  __syncthreads();
  if (tid < 16) {
    float s = 0.f;
#pragma unroll
    for (int j = 0; j < 16; ++j) s += rs[tid][j];
    float z = s + gb[0];
    sgate[tid] = palpha[0] * (1.f / (1.f + expf(-z)));
  }
  __syncthreads();

  // ---- 10. y = LN2(x + gate*out) -> d_out ------------------------------
  float yv[16];
  {
    float g = sgate[row], a1 = 0.f, a2 = 0.f;
#pragma unroll
    for (int j = 0; j < 16; ++j) {
      int c = sub * 16 + j;
      float v = sx[row * CV + c] + g * sout[row * CV + c];
      yv[j] = v; a1 += v; a2 += v * v;
    }
    rs[row][sub] = a1; rq[row][sub] = a2;
  }
  __syncthreads();
  if (tid < 16) {
    float a = 0.f, c2 = 0.f;
#pragma unroll
    for (int j = 0; j < 16; ++j) { a += rs[tid][j]; c2 += rq[tid][j]; }
    float mu = a * (1.f / 256.f);
    smu[tid] = mu; siv[tid] = rsqrtf(c2 * (1.f / 256.f) - mu * mu + 1e-5f);
  }
  __syncthreads();
  {
    float mu = smu[row], iv = siv[row];
    float* dst = yout + gbase + (size_t)row * CV + sub * 16;
#pragma unroll
    for (int j = 0; j < 16; ++j) {
      int c = sub * 16 + j;
      dst[j] = (yv[j] - mu) * iv * n2w[c] + n2b[c];
    }
  }
}

// In-place FFN on y (= d_out): y += gelu(y@W1+b1)@W2+b2.
// 32 rows / block, 8 waves: each B-fragment load feeds TWO WMMAs (row halves),
// halving L2 weight traffic vs. a 16-row tile. y tile staged to LDS with
// async global->LDS (ASYNCcnt) copies.
__global__ __launch_bounds__(256) void ffn_kernel(
    float* __restrict__ y,
    const _Float16* __restrict__ f1h, const float* __restrict__ b1,
    const _Float16* __restrict__ f2h, const float* __restrict__ b2)
{
  __shared__ alignas(32) float    syf[32 * CV];      // y residual (f32)
  __shared__ alignas(32) _Float16 syh[32 * CV];      // y in f16
  __shared__ alignas(32) _Float16 sh[32 * 1024];     // gelu(h) in f16
  const int tid = threadIdx.x, lane = tid & 31, wv = tid >> 5;
  const int col = lane & 15, hi = lane >> 4;
  const size_t g0 = (size_t)blockIdx.x * 32;

  // ---- stage y tile (32x256 f32 = 32KB) via async global->LDS ----------
  {
    const float* gsrc = y + g0 * CV;
    unsigned lbase = (unsigned)(uintptr_t)(&syf[0]);
#pragma unroll
    for (int j = 0; j < 8; ++j) {
      unsigned off = (unsigned)(j * 1024 + tid * 4) * 4u;
      async_g2l_b128(lbase + off, (const void*)(gsrc + j * 1024 + tid * 4));
    }
    wait_async0();
  }
  __syncthreads();
  // f16 copy for WMMA A fragments
#pragma unroll
  for (int j = 0; j < 32; ++j) {
    int idx = j * 256 + tid;
    syh[idx] = (_Float16)syf[idx];
  }
  __syncthreads();

  // ---- GEMM1: 32x1024, K=256 ; +bias, exact gelu, f16 -> sh ------------
  {
    v16h a0[8], a1[8];
#pragma unroll
    for (int k = 0; k < 8; ++k) {
      a0[k] = lda(syh, CV, k * 32);
      a1[k] = lda(syh + 16 * CV, CV, k * 32);
    }
    for (int it = 0; it < 8; ++it) {
      int nt = wv * 8 + it;
      if (it < 7)   // prefetch next n-tile's weight row for this lane
        __builtin_prefetch((const void*)(f1h + (size_t)lane * 1024 + (nt + 1) * 16), 0, 1);
      v8f acc0 = zero8(), acc1 = zero8();
#pragma unroll
      for (int k = 0; k < 8; ++k) {
        v16h bf = ldb(f1h, 1024, k * 32, nt * 16);   // shared by both row halves
        acc0 = wmma16(a0[k], bf, acc0);
        acc1 = wmma16(a1[k], bf, acc1);
      }
      float bc = b1[nt * 16 + col];
#pragma unroll
      for (int r = 0; r < 8; ++r) {
        int m = r + 8 * hi;
        float v0 = acc0[r] + bc;
        float v1 = acc1[r] + bc;
        sh[m * 1024 + nt * 16 + col] =
            (_Float16)(0.5f * v0 * (1.f + erff(v0 * 0.70710678118654752f)));
        sh[(m + 16) * 1024 + nt * 16 + col] =
            (_Float16)(0.5f * v1 * (1.f + erff(v1 * 0.70710678118654752f)));
      }
    }
  }
  __syncthreads();

  // ---- GEMM2: 32x256, K=1024 ; + residual -> y -------------------------
  for (int it = 0; it < 2; ++it) {
    int nt = wv * 2 + it;
    v8f acc0 = zero8(), acc1 = zero8();
    for (int k = 0; k < 32; ++k) {
      v16h af0 = lda(sh, 1024, k * 32);
      v16h af1 = lda(sh + 16 * 1024, 1024, k * 32);
      v16h bf  = ldb(f2h, CV, k * 32, nt * 16);      // shared by both row halves
      acc0 = wmma16(af0, bf, acc0);
      acc1 = wmma16(af1, bf, acc1);
    }
    float bc = b2[nt * 16 + col];
#pragma unroll
    for (int r = 0; r < 8; ++r) {
      int m = r + 8 * hi;
      size_t idx0 = (g0 + m) * CV + nt * 16 + col;
      size_t idx1 = (g0 + m + 16) * CV + nt * 16 + col;
      y[idx0] = syf[m * CV + nt * 16 + col] + acc0[r] + bc;
      y[idx1] = syf[(m + 16) * CV + nt * 16 + col] + acc1[r] + bc;
    }
  }
}

// ---------------------------------------------------------------- launch
extern "C" void kernel_launch(void* const* d_in, const int* in_sizes, int n_in,
                              void* d_out, int out_size, void* d_ws, size_t ws_size,
                              hipStream_t stream)
{
  (void)in_sizes; (void)n_in; (void)out_size; (void)ws_size;
  const float* visual = (const float*)d_in[0];
  const float* text   = (const float*)d_in[1];
  const float* n1w = (const float*)d_in[2];  const float* n1b = (const float*)d_in[3];
  const float* qw  = (const float*)d_in[4];  const float* qb  = (const float*)d_in[5];
  const float* kw  = (const float*)d_in[6];  const float* kb  = (const float*)d_in[7];
  const float* vw  = (const float*)d_in[8];  const float* vb  = (const float*)d_in[9];
  const float* ow  = (const float*)d_in[10]; const float* ob  = (const float*)d_in[11];
  const float* gw  = (const float*)d_in[12]; const float* gb  = (const float*)d_in[13];
  const float* ls  = (const float*)d_in[14]; const float* al  = (const float*)d_in[15];
  const float* n2w = (const float*)d_in[16]; const float* n2b = (const float*)d_in[17];
  const float* f1  = (const float*)d_in[18]; const float* b1  = (const float*)d_in[19];
  const float* f2  = (const float*)d_in[20]; const float* b2  = (const float*)d_in[21];
  float* out = (float*)d_out;

  char* ws = (char*)d_ws;
  size_t off = 0;
  auto alloc = [&](size_t bytes) -> char* {
    char* p = ws + off;
    off = (off + bytes + 255) & ~(size_t)255;
    return p;
  };
  _Float16* qwh = (_Float16*)alloc((size_t)CV * CV * 2);
  _Float16* owh = (_Float16*)alloc((size_t)CV * CV * 2);
  _Float16* f1h = (_Float16*)alloc((size_t)CV * 1024 * 2);
  _Float16* f2h = (_Float16*)alloc((size_t)1024 * CV * 2);
  _Float16* knt = (_Float16*)alloc((size_t)BB * NHEAD * HD * TP1 * 2);
  _Float16* vvv = (_Float16*)alloc((size_t)BB * NHEAD * TP2 * HD * 2);
  float*    bia = (float*)   alloc((size_t)BB * TP1 * 4);

  cvt_f16<<<(CV * CV + 255) / 256, 256, 0, stream>>>(qw, qwh, CV * CV);
  cvt_f16<<<(CV * CV + 255) / 256, 256, 0, stream>>>(ow, owh, CV * CV);
  cvt_f16<<<(CV * 1024 + 255) / 256, 256, 0, stream>>>(f1, f1h, CV * 1024);
  cvt_f16<<<(1024 * CV + 255) / 256, 256, 0, stream>>>(f2, f2h, 1024 * CV);
  text_prep<<<BB * TP2, 256, 0, stream>>>(text, kw, kb, vw, vb, knt, vvv, bia);
  attn_kernel<<<BB * (NN / 16), 256, 0, stream>>>(visual, n1w, n1b, qwh, qb, owh, ob,
                                                  gw, gb, ls, al, n2w, n2b, knt, vvv, bia, out);
  ffn_kernel<<<BB * (NN / 32), 256, 0, stream>>>(out, f1h, b1, f2h, b2);
}
